// TokenChoiceTopKRouter_43035572306001
// MI455X (gfx1250) — matrix operand order repaired and checked by reading, not compiled, not run on hardware
//
#include <hip/hip_runtime.h>
#include <hip/hip_bf16.h>

// ---- CDNA5 WMMA vector types ----
typedef __attribute__((ext_vector_type(16))) _Float16 v16h;
typedef __attribute__((ext_vector_type(8)))  _Float16 h8;
typedef __attribute__((ext_vector_type(4)))  _Float16 h4;
typedef __attribute__((ext_vector_type(8)))  float    v8f;
typedef __attribute__((ext_vector_type(4)))  float    f4;

#define DIM          4096
#define NE           64
#define TOPK         8
#define KCHUNK       512
#define RS           (KCHUNK + 8)     // padded LDS row stride (halves) -> conflict-free b128 reads
#define TOK_PER_BLK  128
#define THREADS      256
#define LSTRIDE      65               // padded logits row stride (floats)

#define LOGITS_BYTES (TOK_PER_BLK * LSTRIDE * 4)   // 33280
#define W_BYTES      (NE * RS * 2)                 // 66560
#define SMEM_BYTES   (LOGITS_BYTES + W_BYTES + NE * 4)

__global__ void zero_counts_kernel(float* counts) {
    counts[threadIdx.x] = 0.0f;
}

__global__ void __launch_bounds__(THREADS)
router_wmma_topk_kernel(const float* __restrict__ x, const float* __restrict__ W,
                        float* __restrict__ out_scores, float* __restrict__ out_idx,
                        float* __restrict__ counts)
{
    extern __shared__ char smem[];
    float*    logitsLds = (float*)smem;
    _Float16* WLds      = (_Float16*)(smem + LOGITS_BYTES);
    float*    hist      = (float*)(smem + LOGITS_BYTES + W_BYTES);

    const int tid  = threadIdx.x;
    const int lane = tid & 31;        // wave32
    const int wv   = tid >> 5;        // 8 waves
    const int lo   = lane & 15;
    const int hiH  = lane >> 4;       // lane half selects K sub-range per ISA layout

    if (tid < NE) hist[tid] = 0.0f;

    const int tokenBase = blockIdx.x * TOK_PER_BLK;
    const float* xrow = x + (size_t)(tokenBase + wv * 16 + lo) * DIM;

    v8f acc[4];
    #pragma unroll
    for (int t = 0; t < 4; ++t) acc[t] = (v8f){};

    const int abase = hiH * 8;    // A frag: lanes 0-15 -> K 0-7,16-23 ; lanes 16-31 -> K 8-15,24-31
    const int koff  = hiH * 8;    // B frag: lanes 0-15 -> K 0-7,16-23 ; lanes 16-31 -> K 8-15,24-31

    // Per-lane LDS base for B fragments; tile offsets become DS immediate offsets.
    const _Float16* bwave = WLds + (size_t)lo * RS + koff;

    for (int kc = 0; kc < DIM; kc += KCHUNK) {
        // Cooperative W chunk: 64 x KCHUNK fp32 -> f16 LDS (padded rows)
        for (int i = 0; i < 32; ++i) {
            int f = i * THREADS + tid;         // 0..8191 float4s
            int e = f >> 7;                    // expert (128 float4 per row chunk)
            int q = f & 127;
            f4 wvv = ((const f4*)(W + (size_t)e * DIM + kc))[q];
            h4 hh;
            hh[0] = (_Float16)wvv[0]; hh[1] = (_Float16)wvv[1];
            hh[2] = (_Float16)wvv[2]; hh[3] = (_Float16)wvv[3];
            *(h4*)(WLds + e * RS + q * 4) = hh;
        }
        __syncthreads();

        if (kc + KCHUNK < DIM)
            __builtin_prefetch(xrow + kc + KCHUNK, 0, 1);   // global_prefetch next x chunk

        for (int k = kc; k < kc + KCHUNK; k += 32) {
            // ---- A fragment: 16 fp32 -> 16 f16 per lane (ISA 16-bit A 16x32 layout) ----
            f4 a0 = *(const f4*)(xrow + k + abase);
            f4 a1 = *(const f4*)(xrow + k + abase + 4);
            f4 a2 = *(const f4*)(xrow + k + abase + 16);
            f4 a3 = *(const f4*)(xrow + k + abase + 20);
            v16h A;
            #pragma unroll
            for (int j = 0; j < 4; ++j) {
                A[j]      = (_Float16)a0[j];
                A[4 + j]  = (_Float16)a1[j];
                A[8 + j]  = (_Float16)a2[j];
                A[12 + j] = (_Float16)a3[j];
            }

            // ---- Batch all 8 B-fragment DS loads (one base + immediate offsets),
            //      so the 4 WMMAs below pipeline instead of draining DScnt each time ----
            const _Float16* bk = bwave + (k - kc);
            h8 b[8];
            #pragma unroll
            for (int t = 0; t < 4; ++t) {
                b[2 * t]     = *(const h8*)(bk + t * 16 * RS);      // ds_load_b128 offset: t*16640
                b[2 * t + 1] = *(const h8*)(bk + t * 16 * RS + 16); // ds_load_b128 offset: +32
            }

            // ---- 4 back-to-back WMMAs (N tiles of 16 experts) ----
            #pragma unroll
            for (int t = 0; t < 4; ++t) {
                v16h B;
                #pragma unroll
                for (int j = 0; j < 8; ++j) { B[j] = b[2 * t][j]; B[8 + j] = b[2 * t + 1][j]; }
                acc[t] = __builtin_amdgcn_wmma_f32_16x16x32_f16(
                    false, A, false, B, (short)0, acc[t], false, false);
            }
        }
        __syncthreads();
    }

    // Spill logits to LDS: C layout -> lane holds (M = r + 8*hiH, N = t*16+lo)
    #pragma unroll
    for (int t = 0; t < 4; ++t) {
        #pragma unroll
        for (int r = 0; r < 8; ++r) {
            logitsLds[(wv * 16 + hiH * 8 + r) * LSTRIDE + t * 16 + lo] = acc[t][r];
        }
    }
    __syncthreads();

    // Top-8 per token (1 thread / token), lowest-index tie-break like lax.top_k
    if (tid < TOK_PER_BLK) {
        const float* row = logitsLds + tid * LSTRIDE;
        const int token = tokenBase + tid;
        unsigned long long used = 0ULL;
        #pragma unroll
        for (int kk = 0; kk < TOPK; ++kk) {
            float best = -3.4e38f;
            int bi = 0;
            for (int e = 0; e < NE; ++e) {
                if ((used >> e) & 1ULL) continue;
                float v = row[e];
                if (v > best) { best = v; bi = e; }
            }
            used |= 1ULL << bi;
            out_scores[token * TOPK + kk] = 1.0f / (1.0f + __expf(-best));
            out_idx[token * TOPK + kk]    = (float)bi;
            atomicAdd(&hist[bi], 1.0f);    // ds_add_f32
        }
    }
    __syncthreads();

    if (tid < NE) atomicAdd(&counts[tid], hist[tid]);   // one global atomic per expert per block
}

extern "C" void kernel_launch(void* const* d_in, const int* in_sizes, int n_in,
                              void* d_out, int out_size, void* d_ws, size_t ws_size,
                              hipStream_t stream) {
    const float* x = (const float*)d_in[0];   // [16384, 4096]
    const float* W = (const float*)d_in[1];   // [64, 4096]
    float* out = (float*)d_out;
    const int n_tokens = in_sizes[0] / DIM;   // 16384

    float* out_scores = out;                              // [tokens, 8]
    float* out_idx    = out + (size_t)n_tokens * TOPK;    // [tokens, 8] (indices as float)
    float* counts     = out + (size_t)2 * n_tokens * TOPK;// [64]

    zero_counts_kernel<<<1, NE, 0, stream>>>(counts);
    router_wmma_topk_kernel<<<n_tokens / TOK_PER_BLK, THREADS, SMEM_BYTES, stream>>>(
        x, W, out_scores, out_idx, counts);
}
